// CombinedLSTMWithStatic2Hop_32375463477764
// MI455X (gfx1250) — matrix operand order repaired and checked by reading, not compiled, not run on hardware
//
#include <hip/hip_runtime.h>
#include <hip/hip_bf16.h>

typedef __bf16 bf16t;
typedef __attribute__((ext_vector_type(4)))  __bf16 v4bf;
typedef __attribute__((ext_vector_type(8)))  __bf16 v8bf;
typedef __attribute__((ext_vector_type(16))) __bf16 v16bf;
typedef __attribute__((ext_vector_type(8)))  float  v8f;
typedef __attribute__((ext_vector_type(4)))  int    v4i;

#define BB 8
#define TT 168
#define NN 1024
#define FDIM 16
#define FS 32
#define FF 8
#define DL 128
#define DG 128
#define HEADS 4
#define EDG 16384
#define BN (BB*NN)          // 8192
#define BEDG (BB*EDG)       // 131072
#define ETOT (BEDG+BN)      // 139264
#define GOUT 512            // 4*DL gate width == HEADS*DG
#define SL 168              // lstm padded K stride (K=144 -> 160 wmma -> 168 pad)
#define SF 392              // fusion padded K stride (K=384)
#define SG 136              // gat padded K stride (K=128)
#define DYNE (BB*TT*NN*FDIM) // 22020096

// ---------------- CDNA5 async global->LDS (ASYNCcnt) with sync fallback ----------------
#if defined(__has_builtin)
# if __has_builtin(__builtin_amdgcn_global_load_async_to_lds_b128)
#  define USE_ASYNC_LDS 1
# endif
#endif

#define AS1 __attribute__((address_space(1)))
#define AS3 __attribute__((address_space(3)))

__device__ __forceinline__ void async_copy16(void* lds_dst, const void* gsrc){
#ifdef USE_ASYNC_LDS
  __builtin_amdgcn_global_load_async_to_lds_b128((AS1 v4i*)gsrc, (AS3 v4i*)lds_dst, 0, 0);
#else
  *(uint4*)lds_dst = *(const uint4*)gsrc;
#endif
}
__device__ __forceinline__ void wait_async(){
#ifdef USE_ASYNC_LDS
# if __has_builtin(__builtin_amdgcn_s_wait_asynccnt)
  __builtin_amdgcn_s_wait_asynccnt(0);
# else
  asm volatile("s_wait_asynccnt 0x0" ::: "memory");
# endif
#endif
}

__device__ __forceinline__ float sigf(float x){ return 1.0f/(1.0f+__expf(-x)); }
__device__ __forceinline__ float tanhfast(float x){ return 2.0f/(1.0f+__expf(-2.0f*x)) - 1.0f; }

// ---- WMMA fragment loaders (wave32; contiguous K per lane half, same order for A and B) ----
// A (16x32, 16-bit): lanes 0-15 -> M=lane, K=k0..k0+15
//                    lanes 16-31 -> M=lane-16, K=k0+16..k0+31
__device__ __forceinline__ v16bf load_a_frag(const bf16t* base, int row0, int k0, int stride){
  int lane = threadIdx.x & 31;
  int m = lane & 15;
  int kb = k0 + ((lane & 16) ? 16 : 0);
  const bf16t* p = base + (size_t)(row0 + m) * stride + kb;
  v8bf lo = *(const v8bf*)p;
  v8bf hi = *(const v8bf*)(p + 8);
  return __builtin_shufflevector(lo, hi, 0,1,2,3,4,5,6,7,8,9,10,11,12,13,14,15);
}
// B (32x16, 16-bit): lanes 0-15 -> N=n0+lane, K=k0..k0+15 ; lanes 16-31 -> N=n0+lane-16, K=k0+16..k0+31
// weights stored [n][k] so the 16 K values are contiguous per lane.
__device__ __forceinline__ v16bf load_b_frag(const bf16t* base, int k0, int n0, int stride){
  int lane = threadIdx.x & 31;
  int n = n0 + (lane & 15);
  int kb = k0 + ((lane & 16) ? 16 : 0);
  const bf16t* p = base + (size_t)n * stride + kb;
  v8bf lo = *(const v8bf*)p;
  v8bf hi = *(const v8bf*)(p + 8);
  return __builtin_shufflevector(lo, hi, 0,1,2,3,4,5,6,7,8,9,10,11,12,13,14,15);
}

// ---------------- weight / input prep ----------------
__global__ void prep_lstm_w_kernel(const float* __restrict__ Wih, const float* __restrict__ Whh,
                                   const float* __restrict__ bih, const float* __restrict__ bhh,
                                   bf16t* __restrict__ Wc, float* __restrict__ biasL){
  int idx = blockIdx.x*256 + threadIdx.x;
  if (idx < GOUT) biasL[idx] = bih[idx] + bhh[idx];
  if (idx >= GOUT*SL) return;
  int n = idx / SL, k = idx % SL;
  float v = 0.0f;
  if (k < FDIM)            v = Wih[n*FDIM + k];
  else if (k < FDIM + DL)  v = Whh[n*DL + (k - FDIM)];
  Wc[idx] = (bf16t)v;
}

__global__ void prep_pad_w_kernel(const float* __restrict__ W, int K, int stride, int NOUT,
                                  bf16t* __restrict__ out){
  int idx = blockIdx.x*256 + threadIdx.x;
  if (idx >= NOUT*stride) return;
  int n = idx / stride, k = idx % stride;
  out[idx] = (bf16t)((k < K) ? W[(size_t)n*K + k] : 0.0f);
}

__global__ void dyn2bf_kernel(const float* __restrict__ src, bf16t* __restrict__ dst){
  int idx = blockIdx.x*256 + threadIdx.x;     // DYNE/4 threads
  float4 v = ((const float4*)src)[idx];
  v4bf o;
  o[0]=(bf16t)v.x; o[1]=(bf16t)v.y; o[2]=(bf16t)v.z; o[3]=(bf16t)v.w;
  ((v4bf*)dst)[idx] = o;
}

// ---------------- static / forecast MLPs into fusion input ----------------
__global__ void sf_kernel(const float* __restrict__ sta, const float* __restrict__ sWt, const float* __restrict__ sbv,
                          const float* __restrict__ fc,  const float* __restrict__ fWt, const float* __restrict__ fbv,
                          bf16t* __restrict__ Abuf){
  int idx = blockIdx.x*256 + threadIdx.x;
  if (idx >= BN*128) return;
  int bn = idx >> 7, c = idx & 127;
  float s = sbv[c];
  const float* sp = sta + (size_t)bn*FS; const float* wr = sWt + (size_t)c*FS;
  #pragma unroll
  for (int k = 0; k < FS; ++k) s += sp[k]*wr[k];
  s = fmaxf(s, 0.0f);
  float f = fbv[c];
  const float* fp = fc + (size_t)bn*FF; const float* wf = fWt + (size_t)c*FF;
  #pragma unroll
  for (int k = 0; k < FF; ++k) f += fp[k]*wf[k];
  f = fmaxf(f, 0.0f);
  bf16t* row = Abuf + (size_t)bn*SF;
  row[128 + c] = (bf16t)s;
  row[256 + c] = (bf16t)f;
  if (c < 8) row[384 + c] = (bf16t)0.0f;   // pad K 384..391
}

// ---------------- LSTM: 64 blocks x 128 sequences ----------------
// Wave w owns gate-column slice [w*16, w*16+16) of each of the 4 gates for ALL 128 rows:
//  - 20 B-fragments (weights) live in VGPRs for the whole 168-step recurrence
//  - cell state c lives in VGPRs
//  - activations [x_t | h] double-buffered in LDS (one barrier per step)
//  - x_t prefetched one step ahead with async global->LDS (ASYNCcnt)
__global__ void __launch_bounds__(256) lstm_kernel(const bf16t* __restrict__ xb,
                                                   const bf16t* __restrict__ Wc,
                                                   const float* __restrict__ biasL,
                                                   bf16t* __restrict__ Abuf){
  extern __shared__ char smem[];
  bf16t* buf0 = (bf16t*)smem;                  // [128][SL]
  bf16t* buf1 = buf0 + (size_t)128*SL;         // [128][SL]
  const int tid = threadIdx.x;
  const int m0 = blockIdx.x * 128;

  { // zero both activation buffers (h0 = 0, K-pad = 0)
    uint4* d = (uint4*)smem;
    const int n16 = 2*128*SL*2/16;
    uint4 z; z.x = z.y = z.z = z.w = 0u;
    for (int i = tid; i < n16; i += 256) d[i] = z;
  }
  __syncthreads();

  const int wid = tid >> 5, lane = tid & 31;

  // weight fragments: resident in registers for all 168 steps
  v16bf bfr[4][5];
  #pragma unroll
  for (int g = 0; g < 4; ++g)
    #pragma unroll
    for (int kt = 0; kt < 5; ++kt)
      bfr[g][kt] = load_b_frag(Wc, kt*32, g*128 + wid*16, SL);

  float bb[4];
  #pragma unroll
  for (int g = 0; g < 4; ++g) bb[g] = biasL[g*128 + wid*16 + (lane & 15)];

  v8f cc[8];
  #pragma unroll
  for (int mt = 0; mt < 8; ++mt)
    #pragma unroll
    for (int q = 0; q < 8; ++q) cc[mt][q] = 0.0f;

  // x_t staging: 2 threads per row, 8 contiguous bf16 (16B) each
  const int xrow = tid >> 1, xhalf = tid & 1;
  const int bn_x = m0 + xrow;
  const int b_x = bn_x >> 10, n_x = bn_x & 1023;
  const bf16t* xsrc = xb + (((size_t)b_x*TT)*NN + n_x)*FDIM + xhalf*8;
  const int xoff = xrow*SL + xhalf*8;

  // prologue: prefetch x(0) into buf0
  async_copy16(buf0 + xoff, xsrc);

  const int hcol = 16 + wid*16 + (lane & 15);
  const int rsel = (lane & 16) ? 8 : 0;

  for (int t = 0; t < TT; ++t){
    bf16t* bufR = (t & 1) ? buf1 : buf0;
    bf16t* bufW = (t & 1) ? buf0 : buf1;
    wait_async();
    __syncthreads();
    if (t + 1 < TT) async_copy16(bufW + xoff, xsrc + (size_t)(t+1)*NN*FDIM);

    #pragma unroll
    for (int mt = 0; mt < 8; ++mt){
      v16bf af[5];
      #pragma unroll
      for (int kt = 0; kt < 5; ++kt) af[kt] = load_a_frag(bufR, mt*16, kt*32, SL);
      v8f acc[4];
      #pragma unroll
      for (int g = 0; g < 4; ++g){
        v8f a;
        #pragma unroll
        for (int q = 0; q < 8; ++q) a[q] = 0.0f;
        #pragma unroll
        for (int kt = 0; kt < 5; ++kt)
          a = __builtin_amdgcn_wmma_f32_16x16x32_bf16(false, af[kt], false, bfr[g][kt],
                                                      (short)0, a, false, false);
        acc[g] = a;
      }
      #pragma unroll
      for (int r = 0; r < 8; ++r){
        float iv = sigf(acc[0][r] + bb[0]);
        float fv = sigf(acc[1][r] + bb[1]);
        float gv = tanhfast(acc[2][r] + bb[2]);
        float ov = sigf(acc[3][r] + bb[3]);
        float cn = fv * cc[mt][r] + iv * gv;
        cc[mt][r] = cn;
        float hv = ov * tanhfast(cn);
        int rr = mt*16 + r + rsel;
        bufW[(size_t)rr*SL + hcol] = (bf16t)hv;
      }
    }
  }
  __syncthreads();
  // final h is in buf[TT&1] == buf0; write z into fusion input cols 0..127 as bf16
  for (int i = tid; i < 128*16; i += 256){
    int row = i >> 4, seg = i & 15;
    v8bf v = *(const v8bf*)(buf0 + (size_t)row*SL + 16 + seg*8);
    *(v8bf*)(Abuf + (size_t)(m0+row)*SF + seg*8) = v;
  }
}

// ---------------- generic LDS-tiled bf16 WMMA GEMM: out = act(A @ W^T + bias) ----------------
__global__ void __launch_bounds__(256) gemm_kernel(const bf16t* __restrict__ A, int lda,
                                                   const bf16t* __restrict__ W, int ldw, int K,
                                                   const float* __restrict__ bias, int actrelu,
                                                   float* __restrict__ outF, int ldo,
                                                   bf16t* __restrict__ outB, int ldob){
  extern __shared__ char smem[];
  bf16t* sW = (bf16t*)smem;            // [128][ldw]
  bf16t* sA = sW + (size_t)128*ldw;    // [128][lda]
  const int tid = threadIdx.x;
  const int m0 = blockIdx.x * 128;
  const int n0 = blockIdx.y * 128;
  {
    uint4* d = (uint4*)sW; const uint4* s = (const uint4*)(W + (size_t)n0*ldw);
    int n16 = 128*ldw*2/16;
    for (int i = tid; i < n16; i += 256) async_copy16(d + i, s + i);
  }
  {
    uint4* d = (uint4*)sA; const uint4* s = (const uint4*)(A + (size_t)m0*lda);
    int n16 = 128*lda*2/16;
    for (int i = tid; i < n16; i += 256) async_copy16(d + i, s + i);
  }
  wait_async();
  __syncthreads();
  const int wid = tid >> 5, lane = tid & 31;
  const int row0 = wid * 16;
  const int nkt = K >> 5;
  v16bf af[12];
  for (int kt = 0; kt < nkt; ++kt) af[kt] = load_a_frag(sA, row0, kt*32, lda);
  for (int j = 0; j < 8; ++j){
    v8f acc;
    #pragma unroll
    for (int q = 0; q < 8; ++q) acc[q] = 0.0f;
    for (int kt = 0; kt < nkt; ++kt){
      v16bf bfrag = load_b_frag(sW, kt*32, j*16, ldw);
      acc = __builtin_amdgcn_wmma_f32_16x16x32_bf16(false, af[kt], false, bfrag,
                                                    (short)0, acc, false, false);
    }
    int col = n0 + j*16 + (lane & 15);
    float bv = bias ? bias[col] : 0.0f;
    #pragma unroll
    for (int r = 0; r < 8; ++r){
      float v = acc[r] + bv;
      if (actrelu) v = fmaxf(v, 0.0f);
      int rr = row0 + r + ((lane & 16) ? 8 : 0);
      outF[(size_t)(m0+rr)*ldo + col] = v;
      if (outB) outB[(size_t)(m0+rr)*ldob + col] = (bf16t)v;
    }
  }
}

// ---------------- graph: edge list + self loops ----------------
__global__ void edges_kernel(const int* __restrict__ eidx, const float* __restrict__ ew,
                             int* __restrict__ srcL, int* __restrict__ dstL, float* __restrict__ eaL,
                             float* __restrict__ deg, float* __restrict__ easum){
  int e = blockIdx.x*256 + threadIdx.x;
  if (e >= BEDG) return;
  int b = e >> 14, i = e & (EDG-1);
  int s = eidx[i]       + b*NN;
  int d = eidx[EDG + i] + b*NN;
  float w = ew[i];
  srcL[e] = s; dstL[e] = d; eaL[e] = w;
  atomicAdd(&deg[d], 1.0f);
  atomicAdd(&easum[d], w);
}

__global__ void loops_kernel(int* __restrict__ srcL, int* __restrict__ dstL, float* __restrict__ eaL,
                             const float* __restrict__ deg, const float* __restrict__ easum){
  int bn = blockIdx.x*256 + threadIdx.x;
  if (bn >= BN) return;
  srcL[BEDG + bn] = bn; dstL[BEDG + bn] = bn;
  eaL[BEDG + bn] = easum[bn] / fmaxf(deg[bn], 1.0f);
}

__global__ void wedot_kernel(const float* __restrict__ We, const float* __restrict__ ae,
                             float* __restrict__ wedot){
  int h = threadIdx.x;
  if (h >= HEADS) return;
  float s = 0.0f;
  for (int c = 0; c < DG; ++c) s += We[h*DG + c] * ae[h*DG + c];
  wedot[h] = s;
}

__global__ void asad_kernel(const float* __restrict__ xl, const float* __restrict__ asrc,
                            const float* __restrict__ adst,
                            float* __restrict__ asb, float* __restrict__ adb){
  int idx = blockIdx.x*256 + threadIdx.x;
  if (idx >= BN*HEADS) return;
  int bn = idx >> 2, h = idx & 3;
  const float* xr = xl + (size_t)bn*GOUT + h*DG;
  float s = 0.0f, d = 0.0f;
  for (int c = 0; c < DG; ++c){ float v = xr[c]; s += v*asrc[h*DG+c]; d += v*adst[h*DG+c]; }
  asb[idx] = s; adb[idx] = d;
}

__global__ void att_kernel(const int* __restrict__ srcL, const int* __restrict__ dstL,
                           const float* __restrict__ eaL,
                           const float* __restrict__ asb, const float* __restrict__ adb,
                           const float* __restrict__ wedot,
                           float* __restrict__ exb, float* __restrict__ den){
  int idx = blockIdx.x*256 + threadIdx.x;
  if (idx >= ETOT*HEADS) return;
  int e = idx >> 2, h = idx & 3;
  float a = asb[srcL[e]*HEADS + h] + adb[dstL[e]*HEADS + h] + eaL[e]*wedot[h];
  a = (a > 0.0f) ? a : 0.2f*a;          // LeakyReLU(0.2)
  float ex = __expf(a);                 // softmax without max-shift (identical ratio)
  exb[idx] = ex;
  atomicAdd(&den[dstL[e]*HEADS + h], ex);
}

__global__ void attnorm_kernel(const int* __restrict__ dstL, float* __restrict__ exb,
                               const float* __restrict__ den){
  int idx = blockIdx.x*256 + threadIdx.x;
  if (idx >= ETOT*HEADS) return;
  int e = idx >> 2, h = idx & 3;
  exb[idx] = exb[idx] / den[dstL[e]*HEADS + h];
}

__global__ void agg_kernel(const int* __restrict__ srcL, const int* __restrict__ dstL,
                           const float* __restrict__ exb, const float* __restrict__ xl,
                           float* __restrict__ out4){
  int idx = blockIdx.x*256 + threadIdx.x;   // < ETOT*512 = 71.3M < 2^31
  int e = idx >> 9, r = idx & 511, h = r >> 7;
  atomicAdd(&out4[(size_t)dstL[e]*GOUT + r], exb[e*HEADS + h] * xl[(size_t)srcL[e]*GOUT + r]);
}

// ---------------- residual + (relu) + LayerNorm, one wave32 per row ----------------
__global__ void __launch_bounds__(256) node_kernel(float* __restrict__ h, const float* __restrict__ out4,
                                                   const float* __restrict__ gbias,
                                                   const float* __restrict__ lng, const float* __restrict__ lnb,
                                                   int relu_delta, bf16t* __restrict__ hb){
  int wid = threadIdx.x >> 5, lane = threadIdx.x & 31;
  int bn = blockIdx.x*8 + wid;
  if (bn >= BN) return;
  float x[4]; float ssum = 0.0f;
  #pragma unroll
  for (int q = 0; q < 4; ++q){
    int c = lane*4 + q;
    const float* o = out4 + (size_t)bn*GOUT + c;
    float d = 0.25f*(o[0] + o[128] + o[256] + o[384]) + gbias[c];  // mean over heads + bias
    if (relu_delta) d = fmaxf(d, 0.0f);
    x[q] = h[(size_t)bn*DG + c] + d;
    ssum += x[q];
  }
  #pragma unroll
  for (int off = 16; off >= 1; off >>= 1) ssum += __shfl_xor(ssum, off, 32);
  float mu = ssum * (1.0f/128.0f);
  float vs = 0.0f;
  #pragma unroll
  for (int q = 0; q < 4; ++q){ float dd = x[q] - mu; vs += dd*dd; }
  #pragma unroll
  for (int off = 16; off >= 1; off >>= 1) vs += __shfl_xor(vs, off, 32);
  float inv = rsqrtf(vs*(1.0f/128.0f) + 1e-5f);
  #pragma unroll
  for (int q = 0; q < 4; ++q){
    int c = lane*4 + q;
    float y = (x[q] - mu)*inv*lng[c] + lnb[c];
    h[(size_t)bn*DG + c] = y;
    hb[(size_t)bn*SG + c] = (bf16t)y;
  }
}

__global__ void out_kernel(const float* __restrict__ h, const float* __restrict__ oW,
                           const float* __restrict__ ob, float* __restrict__ out){
  int bn = blockIdx.x*256 + threadIdx.x;
  if (bn >= BN) return;
  float s = ob[0];
  for (int c = 0; c < DG; ++c) s += h[(size_t)bn*DG + c]*oW[c];
  out[bn] = s;
}

// =======================================================================================
extern "C" void kernel_launch(void* const* d_in, const int* in_sizes, int n_in,
                              void* d_out, int out_size, void* d_ws, size_t ws_size,
                              hipStream_t stream) {
  (void)in_sizes; (void)n_in; (void)out_size; (void)ws_size;
  // inputs in setup_inputs() dict order
  const float* dyn  = (const float*)d_in[0];
  const float* fcst = (const float*)d_in[1];
  const float* sta  = (const float*)d_in[2];
  const int*   eidx = (const int*)  d_in[3];
  const float* ew   = (const float*)d_in[4];
  const float* lWih = (const float*)d_in[5];
  const float* lWhh = (const float*)d_in[6];
  const float* lbih = (const float*)d_in[7];
  const float* lbhh = (const float*)d_in[8];
  const float* sWt  = (const float*)d_in[9];
  const float* sbv  = (const float*)d_in[10];
  const float* fWt  = (const float*)d_in[11];
  const float* fbv  = (const float*)d_in[12];
  const float* uW   = (const float*)d_in[13];
  const float* ub   = (const float*)d_in[14];
  const float* g0W  = (const float*)d_in[15];
  const float* g0as = (const float*)d_in[16];
  const float* g0ad = (const float*)d_in[17];
  const float* g0ae = (const float*)d_in[18];
  const float* g0We = (const float*)d_in[19];
  const float* g0b  = (const float*)d_in[20];
  const float* g1W  = (const float*)d_in[21];
  const float* g1as = (const float*)d_in[22];
  const float* g1ad = (const float*)d_in[23];
  const float* g1ae = (const float*)d_in[24];
  const float* g1We = (const float*)d_in[25];
  const float* g1b  = (const float*)d_in[26];
  const float* ln0g = (const float*)d_in[27];
  const float* ln0b = (const float*)d_in[28];
  const float* ln1g = (const float*)d_in[29];
  const float* ln1b = (const float*)d_in[30];
  const float* oW   = (const float*)d_in[31];
  const float* ob   = (const float*)d_in[32];
  float* outp = (float*)d_out;

  // workspace bump allocator (256B aligned)
  size_t off = 0;
  auto alloc = [&](size_t bytes) -> char* {
    char* p = (char*)d_ws + off;
    off += (bytes + 255) & ~(size_t)255;
    return p;
  };
  bf16t* Wc    = (bf16t*)alloc((size_t)GOUT*SL*2);
  float* biasL = (float*)alloc(GOUT*4);
  bf16t* WbF   = (bf16t*)alloc((size_t)128*SF*2);
  bf16t* Wb0   = (bf16t*)alloc((size_t)GOUT*SG*2);
  bf16t* Wb1   = (bf16t*)alloc((size_t)GOUT*SG*2);
  bf16t* xbb   = (bf16t*)alloc((size_t)DYNE*2);
  bf16t* Abuf  = (bf16t*)alloc((size_t)BN*SF*2);
  float* hbuf  = (float*)alloc((size_t)BN*DG*4);
  bf16t* hb    = (bf16t*)alloc((size_t)BN*SG*2);
  float* xl    = (float*)alloc((size_t)BN*GOUT*4);
  float* out4  = (float*)alloc((size_t)BN*GOUT*4);
  float* asb   = (float*)alloc((size_t)BN*HEADS*4);
  float* adb   = (float*)alloc((size_t)BN*HEADS*4);
  float* den   = (float*)alloc((size_t)BN*HEADS*4);
  float* deg   = (float*)alloc((size_t)BN*4);
  float* easum = (float*)alloc((size_t)BN*4);
  int*   srcL  = (int*)  alloc((size_t)ETOT*4);
  int*   dstL  = (int*)  alloc((size_t)ETOT*4);
  float* eaL   = (float*)alloc((size_t)ETOT*4);
  float* exb   = (float*)alloc((size_t)ETOT*HEADS*4);
  float* wedot = (float*)alloc(HEADS*4);

  // ---- weight / input prep ----
  hipLaunchKernelGGL(prep_lstm_w_kernel, dim3((GOUT*SL+255)/256), dim3(256), 0, stream,
                     lWih, lWhh, lbih, lbhh, Wc, biasL);
  hipLaunchKernelGGL(prep_pad_w_kernel, dim3((128*SF+255)/256), dim3(256), 0, stream,
                     uW, 384, SF, 128, WbF);
  hipLaunchKernelGGL(prep_pad_w_kernel, dim3((GOUT*SG+255)/256), dim3(256), 0, stream,
                     g0W, 128, SG, GOUT, Wb0);
  hipLaunchKernelGGL(prep_pad_w_kernel, dim3((GOUT*SG+255)/256), dim3(256), 0, stream,
                     g1W, 128, SG, GOUT, Wb1);
  hipLaunchKernelGGL(dyn2bf_kernel, dim3(DYNE/4/256), dim3(256), 0, stream, dyn, xbb);

  // ---- static/forecast MLPs -> fusion input cols 128..383 (+pad) ----
  hipLaunchKernelGGL(sf_kernel, dim3(BN*128/256), dim3(256), 0, stream,
                     sta, sWt, sbv, fcst, fWt, fbv, Abuf);

  // ---- LSTM (WMMA, weights register-resident, x_t async-prefetched) ----
  const int lstm_shmem = 2 * 128 * SL * 2;                   // 86016 B
  (void)hipFuncSetAttribute((const void*)lstm_kernel,
                            hipFuncAttributeMaxDynamicSharedMemorySize, lstm_shmem);
  hipLaunchKernelGGL(lstm_kernel, dim3(BN/128), dim3(256), lstm_shmem, stream,
                     xbb, Wc, biasL, Abuf);

  // ---- fusion GEMM: h = relu([z,s,f] @ uW^T + ub) ----
  const int fus_shmem = 256 * SF * 2;                        // 200704 B
  (void)hipFuncSetAttribute((const void*)gemm_kernel,
                            hipFuncAttributeMaxDynamicSharedMemorySize, fus_shmem);
  hipLaunchKernelGGL(gemm_kernel, dim3(BN/128, 1), dim3(256), fus_shmem, stream,
                     Abuf, SF, WbF, SF, 384, ub, 1, hbuf, DG, hb, SG);

  // ---- build batched edge list + mean self-loop attrs ----
  (void)hipMemsetAsync(deg, 0, (size_t)BN*4, stream);
  (void)hipMemsetAsync(easum, 0, (size_t)BN*4, stream);
  hipLaunchKernelGGL(edges_kernel, dim3(BEDG/256), dim3(256), 0, stream,
                     eidx, ew, srcL, dstL, eaL, deg, easum);
  hipLaunchKernelGGL(loops_kernel, dim3(BN/256), dim3(256), 0, stream,
                     srcL, dstL, eaL, deg, easum);

  const int gat_shmem = 256 * SG * 2;                        // 69632 B
  const int nEH = (ETOT*HEADS + 255)/256;
  const int nAgg = ETOT*GOUT/256;

  // ================= hop 0 =================
  (void)hipFuncSetAttribute((const void*)gemm_kernel,
                            hipFuncAttributeMaxDynamicSharedMemorySize, gat_shmem);
  hipLaunchKernelGGL(gemm_kernel, dim3(BN/128, GOUT/128), dim3(256), gat_shmem, stream,
                     hb, SG, Wb0, SG, 128, (const float*)nullptr, 0, xl, GOUT, (bf16t*)nullptr, 0);
  hipLaunchKernelGGL(wedot_kernel, dim3(1), dim3(32), 0, stream, g0We, g0ae, wedot);
  hipLaunchKernelGGL(asad_kernel, dim3(BN*HEADS/256), dim3(256), 0, stream,
                     xl, g0as, g0ad, asb, adb);
  (void)hipMemsetAsync(den, 0, (size_t)BN*HEADS*4, stream);
  (void)hipMemsetAsync(out4, 0, (size_t)BN*GOUT*4, stream);
  hipLaunchKernelGGL(att_kernel, dim3(nEH), dim3(256), 0, stream,
                     srcL, dstL, eaL, asb, adb, wedot, exb, den);
  hipLaunchKernelGGL(attnorm_kernel, dim3(nEH), dim3(256), 0, stream, dstL, exb, den);
  hipLaunchKernelGGL(agg_kernel, dim3(nAgg), dim3(256), 0, stream, srcL, dstL, exb, xl, out4);
  hipLaunchKernelGGL(node_kernel, dim3(BN/8), dim3(256), 0, stream,
                     hbuf, out4, g0b, ln0g, ln0b, 1, hb);

  // ================= hop 1 =================
  (void)hipFuncSetAttribute((const void*)gemm_kernel,
                            hipFuncAttributeMaxDynamicSharedMemorySize, gat_shmem);
  hipLaunchKernelGGL(gemm_kernel, dim3(BN/128, GOUT/128), dim3(256), gat_shmem, stream,
                     hb, SG, Wb1, SG, 128, (const float*)nullptr, 0, xl, GOUT, (bf16t*)nullptr, 0);
  hipLaunchKernelGGL(wedot_kernel, dim3(1), dim3(32), 0, stream, g1We, g1ae, wedot);
  hipLaunchKernelGGL(asad_kernel, dim3(BN*HEADS/256), dim3(256), 0, stream,
                     xl, g1as, g1ad, asb, adb);
  (void)hipMemsetAsync(den, 0, (size_t)BN*HEADS*4, stream);
  (void)hipMemsetAsync(out4, 0, (size_t)BN*GOUT*4, stream);
  hipLaunchKernelGGL(att_kernel, dim3(nEH), dim3(256), 0, stream,
                     srcL, dstL, eaL, asb, adb, wedot, exb, den);
  hipLaunchKernelGGL(attnorm_kernel, dim3(nEH), dim3(256), 0, stream, dstL, exb, den);
  hipLaunchKernelGGL(agg_kernel, dim3(nAgg), dim3(256), 0, stream, srcL, dstL, exb, xl, out4);
  hipLaunchKernelGGL(node_kernel, dim3(BN/8), dim3(256), 0, stream,
                     hbuf, out4, g1b, ln1g, ln1b, 0, hb);

  // ---- readout ----
  hipLaunchKernelGGL(out_kernel, dim3(BN/256), dim3(256), 0, stream, hbuf, oW, ob, outp);
}
